// LSTM_46505905881681
// MI455X (gfx1250) — compile-verified, matrix operand order checked by reading, not run on hardware
//
#include <hip/hip_runtime.h>
#include <hip/hip_bf16.h>

typedef __attribute__((ext_vector_type(16))) _Float16 v16h;
typedef __attribute__((ext_vector_type(8)))  _Float16 v8h;
typedef __attribute__((ext_vector_type(8)))  float    v8f;

#define B_   64
#define D_   32
#define H_   512
#define T_   1024
#define NWG  64
#define NHPW 8            // hidden units per workgroup
#define ROWS 32           // gate rows per workgroup (4 gates * NHPW)
#define NCHUNK 16         // H / 32 (K-chunks for WMMA f16 K=32)
#define HROW 520          // LDS stage row stride in halves (512 + 8 pad = 1040 B)

// ---------------- math helpers (safe for large |x|) ----------------
__device__ __forceinline__ float sigmoid_(float x) {
    return 1.0f / (1.0f + __expf(-x));
}
__device__ __forceinline__ float tanh_(float x) {
    float e = __expf(-2.0f * fabsf(x));      // in (0,1], never overflows
    float r = (1.0f - e) / (1.0f + e);
    return copysignf(r, x);
}

// ---------------- grid-wide barrier (agent-scope atomics) ----------------
__device__ __forceinline__ void grid_barrier(unsigned* cnt, unsigned* gen, unsigned nwg) {
    __threadfence();          // flush this wave's global stores/atomics to agent scope
    __syncthreads();          // all waves in WG have fenced
    if (threadIdx.x == 0) {
        unsigned g = __hip_atomic_load(gen, __ATOMIC_RELAXED, __HIP_MEMORY_SCOPE_AGENT);
        unsigned a = __hip_atomic_fetch_add(cnt, 1u, __ATOMIC_ACQ_REL, __HIP_MEMORY_SCOPE_AGENT);
        if (a == nwg - 1u) {
            __hip_atomic_store(cnt, 0u, __ATOMIC_RELAXED, __HIP_MEMORY_SCOPE_AGENT);
            __hip_atomic_fetch_add(gen, 1u, __ATOMIC_RELEASE, __HIP_MEMORY_SCOPE_AGENT);
        } else {
            while (__hip_atomic_load(gen, __ATOMIC_ACQUIRE, __HIP_MEMORY_SCOPE_AGENT) == g) {
                __builtin_amdgcn_s_sleep(1);   // fine-grained wake (~64 clk)
            }
        }
    }
    __syncthreads();
}

// ---------------- init: out = conv_b, h0 = 0, sync = 0 ----------------
__global__ __launch_bounds__(256) void init_buffers(float* __restrict__ out,
                                                    const float* __restrict__ conv_b,
                                                    _Float16* __restrict__ h0,
                                                    unsigned* __restrict__ sync) {
    int idx = blockIdx.x * 256 + threadIdx.x;        // 65536 threads
    if (idx < B_ * T_) out[idx] = conv_b[0];
    if (idx < B_ * H_) h0[idx]  = (_Float16)0.0f;
    if (idx < 2)       sync[idx] = 0u;
}

// ---------------- tiled transpose+convert: x[BD][T] f32 -> xT[T][BD] f16 ----------------
__global__ __launch_bounds__(256) void transpose_x(const float* __restrict__ x,
                                                   _Float16* __restrict__ xT) {
    __shared__ float tile[32][33];
    const int tc = blockIdx.x;            // t-tile   (32 tiles)
    const int tr = blockIdx.y;            // bd-tile  (64 tiles)
    const int lx = threadIdx.x;           // 0..31
    const int ly = threadIdx.y;           // 0..7
    #pragma unroll
    for (int i = ly; i < 32; i += 8)
        tile[i][lx] = x[(size_t)(tr * 32 + i) * T_ + tc * 32 + lx];
    __syncthreads();
    #pragma unroll
    for (int i = ly; i < 32; i += 8)
        xT[(size_t)(tc * 32 + i) * (B_ * D_) + tr * 32 + lx] = (_Float16)tile[lx][i];
}

// ---------------- persistent LSTM kernel ----------------
__global__ __launch_bounds__(256, 1) void lstm_persistent(
    const _Float16* __restrict__ xT,     // [T][B][D] f16
    const float*    __restrict__ W_ih,   // [4H][D]
    const float*    __restrict__ W_hh,   // [4H][H]
    const float*    __restrict__ b_ih,   // [4H]
    const float*    __restrict__ b_hh,   // [4H]
    const float*    __restrict__ conv_w, // [H]
    _Float16*       __restrict__ hbuf,   // [2][B][H] f16 (double buffer)
    unsigned*       __restrict__ sync,   // [2] {count, generation}
    float*          __restrict__ out)    // [B][T] pre-seeded with conv_b
{
    __shared__ __align__(16) _Float16 hstage[B_ * HROW];   // staged h (65 KB, padded rows)
    __shared__ float gbuf[ROWS][B_ + 1];                   // gate tiles, padded (8.3 KB)
    __shared__ float cstate[B_][NHPW];                     // cell state (2 KB)

    const int wg   = blockIdx.x;
    const int tid  = threadIdx.x;
    const int lane = tid & 31;
    const int wave = tid >> 5;
    const int m15  = lane & 15;                // A-row / B-col within 16x16 tile
    const int hi   = lane >> 4;
    const int kb   = hi * 8;                   // K sub-base per ISA f16 layout
    const int rg   = wave >> 2;                // row-group: local rows [rg*16, rg*16+16)
    const int b0   = (wave & 3) * 16;          // batch base of this wave's tile
    const int r    = rg * 16 + m15;            // local gate row 0..31
    const int gate = r >> 3;                   // 0=i 1=f 2=g 3=o
    const int jj   = r & 7;
    const int gr   = gate * H_ + wg * NHPW + jj;   // global gate row in [0,4H)

    // zero cell state
    for (int i = tid; i < B_ * NHPW; i += 256) ((float*)cstate)[i] = 0.0f;

    // ---- one-time: pin W_hh slice in VGPRs as packed B-fragments (f16) ----
    v16h bfrag[NCHUNK];
    {
        const float* wrow = W_hh + (size_t)gr * H_;
        #pragma unroll
        for (int c = 0; c < NCHUNK; ++c) {
            v16h f;
            #pragma unroll
            for (int i = 0; i < 8; ++i) {
                f[i]     = (_Float16)wrow[c * 32 + kb + i];
                f[8 + i] = (_Float16)wrow[c * 32 + kb + 16 + i];
            }
            bfrag[c] = f;
        }
    }
    v16h bifrag;                                  // W_ih fragment (single K=32 chunk)
    {
        const float* wirow = W_ih + gr * D_;
        #pragma unroll
        for (int i = 0; i < 8; ++i) {
            bifrag[i]     = (_Float16)wirow[kb + i];
            bifrag[8 + i] = (_Float16)wirow[kb + 16 + i];
        }
    }
    const float biasv = b_ih[gr] + b_hh[gr];

    // combine-phase constants for this thread
    const int   cb  = tid >> 2;                 // batch handled in combine phase
    const int   cq  = tid & 3;                  // quarter -> hidden pair {2q, 2q+1}
    const float cw0 = conv_w[wg * NHPW + 2 * cq];
    const float cw1 = conv_w[wg * NHPW + 2 * cq + 1];
    const unsigned nwg = gridDim.x;

    for (int t = 0; t < T_; ++t) {
        const _Float16* hcur  = hbuf + (size_t)(t & 1) * (B_ * H_);
        _Float16*       hnext = hbuf + (size_t)((t + 1) & 1) * (B_ * H_);

        // ---- stage entire h (64 KB f16) into LDS: bulk, pipelined loads ----
        // 4096 16-byte units; 256 threads x 16 units, in two batches of 8
        #pragma unroll
        for (int half = 0; half < 2; ++half) {
            v8h tmp[8];
            #pragma unroll
            for (int k = 0; k < 8; ++k) {
                int u   = tid + (half * 8 + k) * 256;
                int row = u >> 6;             // 64 x 16B units per 1024B row
                int col = (u & 63) * 8;       // in halves
                tmp[k] = *reinterpret_cast<const v8h*>(hcur + (size_t)row * H_ + col);
            }
            #pragma unroll
            for (int k = 0; k < 8; ++k) {
                int u   = tid + (half * 8 + k) * 256;
                int row = u >> 6;
                int col = (u & 63) * 8;
                *reinterpret_cast<v8h*>(&hstage[row * HROW + col]) = tmp[k];
            }
        }
        // xp fragment loads issued before the sync (independent of LDS stage)
        const _Float16* arow = xT + (size_t)t * (B_ * D_) + (size_t)(b0 + m15) * D_;
        v8h xlo = *reinterpret_cast<const v8h*>(arow + kb);
        v8h xhx = *reinterpret_cast<const v8h*>(arow + kb + 16);
        __syncthreads();

        // two independent accumulator chains: acc0 carries bias + x-projection
        v8f acc0, acc1;
        #pragma unroll
        for (int i = 0; i < 8; ++i) { acc0[i] = biasv; acc1[i] = 0.0f; }

        // fused input projection (chain 0; its loads were issued pre-barrier)
        {
            v16h a = __builtin_shufflevector(xlo, xhx, 0,1,2,3,4,5,6,7,8,9,10,11,12,13,14,15);
            acc0 = __builtin_amdgcn_wmma_f32_16x16x32_f16(false, a, false, bifrag,
                                                          (short)0, acc0, false, false);
        }
        // recurrent GEMM from LDS: even chunks -> chain0, odd chunks -> chain1
        {
            const _Float16* lrow = &hstage[(b0 + m15) * HROW];
            #pragma unroll
            for (int c = 0; c < NCHUNK; c += 2) {
                v8h lo0 = *reinterpret_cast<const v8h*>(lrow + c * 32 + kb);
                v8h hx0 = *reinterpret_cast<const v8h*>(lrow + c * 32 + kb + 16);
                v8h lo1 = *reinterpret_cast<const v8h*>(lrow + (c + 1) * 32 + kb);
                v8h hx1 = *reinterpret_cast<const v8h*>(lrow + (c + 1) * 32 + kb + 16);
                v16h a0 = __builtin_shufflevector(lo0, hx0,
                                                  0,1,2,3,4,5,6,7,8,9,10,11,12,13,14,15);
                v16h a1 = __builtin_shufflevector(lo1, hx1,
                                                  0,1,2,3,4,5,6,7,8,9,10,11,12,13,14,15);
                acc0 = __builtin_amdgcn_wmma_f32_16x16x32_f16(false, a0, false, bfrag[c],
                                                              (short)0, acc0, false, false);
                acc1 = __builtin_amdgcn_wmma_f32_16x16x32_f16(false, a1, false, bfrag[c + 1],
                                                              (short)0, acc1, false, false);
            }
        }

        // merge chains and spill C tile to LDS: row = local gate row, col = batch
        #pragma unroll
        for (int vr = 0; vr < 8; ++vr)
            gbuf[r][b0 + hi * 8 + vr] = acc0[vr] + acc1[vr];
        __syncthreads();

        // elementwise gate math + cell update + conv partial
        {
            float partial = 0.0f;
            unsigned hp = 0u;
            #pragma unroll
            for (int e = 0; e < 2; ++e) {
                int j   = 2 * cq + e;
                float gi = gbuf[j][cb];
                float gf = gbuf[8  + j][cb];
                float gg = gbuf[16 + j][cb];
                float go = gbuf[24 + j][cb];
                float iv = sigmoid_(gi);
                float fv = sigmoid_(gf);
                float gv = tanh_(gg);
                float ov = sigmoid_(go);
                float cn = fv * cstate[cb][j] + iv * gv;
                cstate[cb][j] = cn;
                float hv = ov * tanh_(cn);
                partial += hv * (e ? cw1 : cw0);
                union { _Float16 h; unsigned short u; } pk;
                pk.h = (_Float16)hv;
                hp |= ((unsigned)pk.u) << (16 * e);
            }
            // packed f16 pair store of h_new (4B aligned)
            *reinterpret_cast<unsigned*>(hnext + (size_t)cb * H_ + wg * NHPW + 2 * cq) = hp;

            // reduce conv partial across the 4 threads of this batch (adjacent lanes)
            partial += __shfl_xor(partial, 1, 32);
            partial += __shfl_xor(partial, 2, 32);
            if (cq == 0)
                __hip_atomic_fetch_add(&out[(size_t)cb * T_ + t], partial,
                                       __ATOMIC_RELAXED, __HIP_MEMORY_SCOPE_AGENT);
        }
        // one global barrier per timestep (also protects hstage/gbuf reuse)
        grid_barrier(sync, sync + 1, nwg);
    }
}

// ---------------- host entry ----------------
extern "C" void kernel_launch(void* const* d_in, const int* in_sizes, int n_in,
                              void* d_out, int out_size, void* d_ws, size_t ws_size,
                              hipStream_t stream) {
    const float* x    = (const float*)d_in[0];   // (B, D, T)
    const float* Wih  = (const float*)d_in[1];   // (4H, D)
    const float* Whh  = (const float*)d_in[2];   // (4H, H)
    const float* bih  = (const float*)d_in[3];   // (4H,)
    const float* bhh  = (const float*)d_in[4];   // (4H,)
    const float* cwp  = (const float*)d_in[5];   // (1, H, 1)
    const float* cbp  = (const float*)d_in[6];   // (1,)
    float* out = (float*)d_out;                  // (B, 1, T)

    char* ws = (char*)d_ws;
    unsigned* sync = (unsigned*)ws;                              // 256 B reserved
    _Float16* hbuf = (_Float16*)(ws + 256);                      // 2*64*512*2 = 128 KB
    _Float16* xT   = (_Float16*)(ws + 256 + 2 * B_ * H_ * 2);    // 1024*64*32*2 = 4 MB

    init_buffers<<<256, 256, 0, stream>>>(out, cbp, hbuf, sync);

    dim3 tb(32, 8), tg(T_ / 32, (B_ * D_) / 32);
    transpose_x<<<tg, tb, 0, stream>>>(x, xT);

    lstm_persistent<<<NWG, 256, 0, stream>>>(xT, Wih, Whh, bih, bhh, cwp, hbuf, sync, out);
}
